// SynthesizerDense_86002425135417
// MI455X (gfx1250) — matrix operand order repaired and checked by reading, not compiled, not run on hardware
//
#include <hip/hip_runtime.h>
#include <hip/hip_bf16.h>

// ---------------------------------------------------------------------------
// Synthesizer dense attention for MI455X (gfx1250, wave32, WMMA).
//   b=64, n=300, d=2048, MAX_SEQ_LEN=300
// Pipeline:
//   cvt w1,w2,x -> f16 ; x transposed copy xT[b][d][s] (s padded to 320, zeros)
//   GEMM1: h = relu(x*W1^T + b1)             f16 in, f32 acc, f16 out (32x64/wave)
//   GEMM2: energy = h*W2^T + b2, causal mask f32 out                (32x32/wave)
//   softmax rows -> prob (f16)
//   GEMM3: out = prob * x                    f32 out                (32x32/wave)
// All matrix math uses v_wmma_f32_16x16x32_f16.
// ---------------------------------------------------------------------------

typedef _Float16 v16h __attribute__((ext_vector_type(16)));
typedef _Float16 v8h  __attribute__((ext_vector_type(8)));
typedef float    v8f  __attribute__((ext_vector_type(8)));

#define NEG_INF (-1e30f)

// ---- geometry -------------------------------------------------------------
#define BATCH 64
#define SEQ   300
#define DIM   2048
#define MTOT  (BATCH * SEQ)     // 19200 flat rows, divisible by 64
#define MPAD  19264             // h row allocation (covers b*300+319 reads)
#define SP    320               // padded seq (K mult of 32, tiles of 64)

// ---- workspace offsets (bytes, 256-aligned). Total ~291 MB. ---------------
#define OFF_W1 ((size_t)0)                         // f16 [2048][2048]   8.39 MB
#define OFF_W2 ((size_t)8388608)                   // f16 [320][2048]    1.31 MB
#define OFF_X  ((size_t)9699328)                   // f16 [19264][2048] 78.91 MB
#define OFF_XT ((size_t)88604672)                  // f16 [64][2048][320] 83.9 MB
#define OFF_H  ((size_t)172490752)                 // f16 [19264][2048] 78.91 MB
#define OFF_E  ((size_t)251396096)                 // f32 [64][320][320] 26.2 MB
#define OFF_P  ((size_t)277610496)                 // f16 [64][320][320] 13.1 MB

// ---------------------------------------------------------------------------
// Fragment loaders matching CDNA5 ISA §7.12.2 VGPR layouts.
// A (16x32, M x K), source row-major, K contiguous:
//   lanes 0-15 : row=lane,   K = 0..7  and 16..23
//   lanes16-31 : row=lane-16,K = 8..15 and 24..31
__device__ __forceinline__ v16h load_frag_a(const _Float16* __restrict__ base, int ld) {
  const int lane = threadIdx.x & 31;
  const int row  = lane & 15;
  const int kb   = (lane >> 4) << 3;            // 0 or 8
  const _Float16* q = base + (size_t)row * ld + kb;
  v8h lo = *reinterpret_cast<const v8h*>(q);
  v8h hi = *reinterpret_cast<const v8h*>(q + 16);
  v16h a;
#pragma unroll
  for (int i = 0; i < 8; ++i) { a[i] = lo[i]; a[i + 8] = hi[i]; }
  return a;
}

// B (32x16, K x N), source "column-major" (element (k,n) at base[n*ld + k]):
//   lanes 0-15 : col=lane,   K = 0..7  and 16..23
//   lanes16-31 : col=lane-16,K = 8..15 and 24..31
__device__ __forceinline__ v16h load_frag_b(const _Float16* __restrict__ base, int ld) {
  const int lane = threadIdx.x & 31;
  const int col  = lane & 15;
  const int kb   = (lane >> 4) << 3;            // 0 or 8
  const _Float16* q = base + (size_t)col * ld + kb;
  v8h lo = *reinterpret_cast<const v8h*>(q);
  v8h hi = *reinterpret_cast<const v8h*>(q + 16);
  v16h b;
#pragma unroll
  for (int i = 0; i < 8; ++i) { b[i] = lo[i]; b[i + 8] = hi[i]; }
  return b;
}

#define WMMA_F16(A, B, C) \
  __builtin_amdgcn_wmma_f32_16x16x32_f16(false, (A), false, (B), (short)0, (C), false, false)

// ---------------------------------------------------------------------------
// Conversion kernels
__global__ void __launch_bounds__(256)
cvt_f32_to_f16(const float* __restrict__ src, _Float16* __restrict__ dst, size_t n) {
  size_t i = (size_t)blockIdx.x * blockDim.x + threadIdx.x;
  if (i < n) dst[i] = (_Float16)src[i];
}

__global__ void __launch_bounds__(256)
cvt_w2_pad(const float* __restrict__ w2, _Float16* __restrict__ dst) {
  size_t i = (size_t)blockIdx.x * blockDim.x + threadIdx.x;   // over SP*DIM
  if (i >= (size_t)SP * DIM) return;
  int s = (int)(i >> 11);
  dst[i] = (s < SEQ) ? (_Float16)w2[i] : (_Float16)0.f;
}

// x[b][s][d] (f32) -> xT[b][d][s] (f16), s padded to SP with zeros
__global__ void __launch_bounds__(256)
transpose_x(const float* __restrict__ x, _Float16* __restrict__ xt) {
  __shared__ _Float16 tile[32][33];
  const int b  = blockIdx.z;
  const int s0 = blockIdx.y * 32;       // 10 tiles -> 320
  const int d0 = blockIdx.x * 32;       // 64 tiles -> 2048
  const int tx = threadIdx.x;           // 32
  const int ty = threadIdx.y;           // 8
#pragma unroll
  for (int r = 0; r < 32; r += 8) {
    int s = s0 + ty + r, d = d0 + tx;
    float v = (s < SEQ) ? x[((size_t)b * SEQ + s) * DIM + d] : 0.f;
    tile[ty + r][tx] = (_Float16)v;
  }
  __syncthreads();
#pragma unroll
  for (int r = 0; r < 32; r += 8) {
    int d = d0 + ty + r, s = s0 + tx;
    xt[((size_t)b * DIM + d) * SP + s] = tile[tx][ty + r];
  }
}

// ---------------------------------------------------------------------------
// GEMM1: h[19200][2048] = relu(x[19200][2048] * W1^T + b1), f16 out.
// Block: 256 thr = 8 waves (2 M x 4 N); wave tile 32(M) x 64(N);
// block tile 64 x 256. 8 WMMA per K-step, 2 A-frags + 4 B-frags loaded.
__global__ void __launch_bounds__(256)
gemm1_relu(const _Float16* __restrict__ x, const _Float16* __restrict__ w1,
           const float* __restrict__ b1, _Float16* __restrict__ h) {
  const int w  = threadIdx.x >> 5;
  const int m0 = blockIdx.y * 64 + (w >> 2) * 32;      // 2 waves in M
  const int n0 = blockIdx.x * 256 + (w & 3) * 64;      // 4 waves in N (64 wide)
  v8f acc[2][4] = {};
  const _Float16* ap = x + (size_t)m0 * DIM;
  const _Float16* bp = w1 + (size_t)n0 * DIM;
  for (int k = 0; k < DIM; k += 32) {
    // WGP-scope prefetch two K-steps ahead for the streamed A rows.
    if (k + 64 < DIM) {
      __builtin_prefetch(ap + k + 64, 0, 3);
      __builtin_prefetch(ap + (size_t)16 * DIM + k + 64, 0, 3);
    }
    v16h a0 = load_frag_a(ap + k, DIM);
    v16h a1 = load_frag_a(ap + (size_t)16 * DIM + k, DIM);
    v16h bf[4];
#pragma unroll
    for (int j = 0; j < 4; ++j)
      bf[j] = load_frag_b(bp + (size_t)(16 * j) * DIM + k, DIM);
#pragma unroll
    for (int j = 0; j < 4; ++j) acc[0][j] = WMMA_F16(a0, bf[j], acc[0][j]);
#pragma unroll
    for (int j = 0; j < 4; ++j) acc[1][j] = WMMA_F16(a1, bf[j], acc[1][j]);
  }
  const int lane = threadIdx.x & 31;
  const int cn = lane & 15;
  const int rb = (lane >> 4) * 8;
#pragma unroll
  for (int i = 0; i < 2; ++i)
#pragma unroll
    for (int j = 0; j < 4; ++j) {
      int col = n0 + j * 16 + cn;
      float bias = b1[col];
#pragma unroll
      for (int r = 0; r < 8; ++r) {
        int row = m0 + i * 16 + rb + r;
        float v = acc[i][j][r] + bias;
        h[(size_t)row * DIM + col] = (_Float16)(v > 0.f ? v : 0.f);
      }
    }
}

// ---------------------------------------------------------------------------
// GEMM2: energy[b][m][s] = h_b * W2^T + b2, causal mask, f32 out [64][320][320]
// Block: 128 thr = 4 waves (2 M x 2 N), block tile 64 x 64.
__global__ void __launch_bounds__(128)
gemm2_mask(const _Float16* __restrict__ h, const _Float16* __restrict__ w2,
           const float* __restrict__ b2, float* __restrict__ energy) {
  const int b  = blockIdx.z;
  const int w  = threadIdx.x >> 5;
  const int m0 = blockIdx.y * 64 + (w >> 1) * 32;
  const int s0 = blockIdx.x * 64 + (w & 1) * 32;
  const _Float16* ap = h + ((size_t)b * SEQ + m0) * DIM;
  const _Float16* bp = w2 + (size_t)s0 * DIM;
  v8f acc00 = {}, acc01 = {}, acc10 = {}, acc11 = {};
  for (int k = 0; k < DIM; k += 32) {
    v16h a0 = load_frag_a(ap + k, DIM);
    v16h a1 = load_frag_a(ap + (size_t)16 * DIM + k, DIM);
    v16h f0 = load_frag_b(bp + k, DIM);
    v16h f1 = load_frag_b(bp + (size_t)16 * DIM + k, DIM);
    acc00 = WMMA_F16(a0, f0, acc00);
    acc01 = WMMA_F16(a0, f1, acc01);
    acc10 = WMMA_F16(a1, f0, acc10);
    acc11 = WMMA_F16(a1, f1, acc11);
  }
  const int lane = threadIdx.x & 31;
  const int cn = lane & 15;
  const int rb = (lane >> 4) * 8;
  v8f* accs[2][2] = {{&acc00, &acc01}, {&acc10, &acc11}};
  float* erow = energy + (size_t)b * SP * SP;
#pragma unroll
  for (int i = 0; i < 2; ++i)
#pragma unroll
    for (int j = 0; j < 2; ++j) {
      int s = s0 + j * 16 + cn;
#pragma unroll
      for (int r = 0; r < 8; ++r) {
        int m = m0 + i * 16 + rb + r;
        float v = (s < SEQ && s <= m) ? ((*accs[i][j])[r] + b2[s]) : NEG_INF;
        erow[(size_t)m * SP + s] = v;
      }
    }
}

// ---------------------------------------------------------------------------
// Row softmax over 320 cols: one wave32 per row, 8 rows per 256-thr block.
__global__ void __launch_bounds__(256)
softmax_rows(const float* __restrict__ energy, _Float16* __restrict__ prob) {
  const int row  = blockIdx.x * 8 + (threadIdx.x >> 5);   // [0, 64*320)
  const int lane = threadIdx.x & 31;
  const float* e = energy + (size_t)row * SP;
  float vals[10];
  float mx = -3.4e38f;
#pragma unroll
  for (int i = 0; i < 10; ++i) { vals[i] = e[lane + 32 * i]; mx = fmaxf(mx, vals[i]); }
#pragma unroll
  for (int off = 16; off; off >>= 1) mx = fmaxf(mx, __shfl_xor(mx, off, 32));
  float sum = 0.f;
#pragma unroll
  for (int i = 0; i < 10; ++i) { vals[i] = __expf(vals[i] - mx); sum += vals[i]; }
#pragma unroll
  for (int off = 16; off; off >>= 1) sum += __shfl_xor(sum, off, 32);
  float inv = 1.f / sum;
  _Float16* p = prob + (size_t)row * SP;
#pragma unroll
  for (int i = 0; i < 10; ++i) p[lane + 32 * i] = (_Float16)(vals[i] * inv);
}

// ---------------------------------------------------------------------------
// GEMM3: out[b][m][d] = prob[b][m][s] * x[b][s][d]  (B from transposed xT)
// Block: 128 thr = 4 waves, block tile 64 x 64; K = SP = 320.
__global__ void __launch_bounds__(128)
gemm3_out(const _Float16* __restrict__ prob, const _Float16* __restrict__ xt,
          float* __restrict__ out) {
  const int b  = blockIdx.z;
  const int w  = threadIdx.x >> 5;
  const int m0 = blockIdx.y * 64 + (w >> 1) * 32;
  const int d0 = blockIdx.x * 64 + (w & 1) * 32;
  const _Float16* ap = prob + (size_t)b * SP * SP + (size_t)m0 * SP;
  const _Float16* bp = xt + (size_t)b * DIM * SP + (size_t)d0 * SP;
  v8f acc00 = {}, acc01 = {}, acc10 = {}, acc11 = {};
  for (int k = 0; k < SP; k += 32) {
    v16h a0 = load_frag_a(ap + k, SP);
    v16h a1 = load_frag_a(ap + (size_t)16 * SP + k, SP);
    v16h f0 = load_frag_b(bp + k, SP);
    v16h f1 = load_frag_b(bp + (size_t)16 * SP + k, SP);
    acc00 = WMMA_F16(a0, f0, acc00);
    acc01 = WMMA_F16(a0, f1, acc01);
    acc10 = WMMA_F16(a1, f0, acc10);
    acc11 = WMMA_F16(a1, f1, acc11);
  }
  const int lane = threadIdx.x & 31;
  const int cn = lane & 15;
  const int rb = (lane >> 4) * 8;
  v8f* accs[2][2] = {{&acc00, &acc01}, {&acc10, &acc11}};
#pragma unroll
  for (int i = 0; i < 2; ++i)
#pragma unroll
    for (int j = 0; j < 2; ++j) {
      int d = d0 + j * 16 + cn;
#pragma unroll
      for (int r = 0; r < 8; ++r) {
        int m = m0 + i * 16 + rb + r;
        if (m < SEQ) out[((size_t)b * SEQ + m) * DIM + d] = (*accs[i][j])[r];
      }
    }
}

// ---------------------------------------------------------------------------
extern "C" void kernel_launch(void* const* d_in, const int* in_sizes, int n_in,
                              void* d_out, int out_size, void* d_ws, size_t ws_size,
                              hipStream_t stream) {
  const float* x   = (const float*)d_in[0];
  const float* w1w = (const float*)d_in[1];
  const float* w1b = (const float*)d_in[2];
  const float* w2w = (const float*)d_in[3];
  const float* w2b = (const float*)d_in[4];
  float* out = (float*)d_out;

  char* ws = (char*)d_ws;
  _Float16* w1h = (_Float16*)(ws + OFF_W1);
  _Float16* w2h = (_Float16*)(ws + OFF_W2);
  _Float16* xh  = (_Float16*)(ws + OFF_X);
  _Float16* xth = (_Float16*)(ws + OFF_XT);
  _Float16* hh  = (_Float16*)(ws + OFF_H);
  float*    eng = (float*)   (ws + OFF_E);
  _Float16* prb = (_Float16*)(ws + OFF_P);

  // conversions / transposed copy
  {
    size_t n = (size_t)DIM * DIM;
    cvt_f32_to_f16<<<(unsigned)((n + 255) / 256), 256, 0, stream>>>(w1w, w1h, n);
  }
  {
    size_t n = (size_t)MTOT * DIM;
    cvt_f32_to_f16<<<(unsigned)((n + 255) / 256), 256, 0, stream>>>(x, xh, n);
  }
  {
    size_t n = (size_t)SP * DIM;
    cvt_w2_pad<<<(unsigned)((n + 255) / 256), 256, 0, stream>>>(w2w, w2h);
  }
  transpose_x<<<dim3(DIM / 32, SP / 32, BATCH), dim3(32, 8), 0, stream>>>(x, xth);

  // GEMM1: 19200 x 2048 x 2048, block tile 64x256
  gemm1_relu<<<dim3(DIM / 256, MTOT / 64), 256, 0, stream>>>(xh, w1h, w1b, hh);

  // GEMM2 + mask: per batch 320 x 320 x 2048, block tile 64x64
  gemm2_mask<<<dim3(SP / 64, SP / 64, BATCH), 128, 0, stream>>>(hh, w2h, w2b, eng);

  // softmax: one wave per row
  softmax_rows<<<(BATCH * SP) / 8, 256, 0, stream>>>(eng, prb);

  // GEMM3: per batch 320 x 2048 x 320, block tile 64x64
  gemm3_out<<<dim3(DIM / 64, SP / 64, BATCH), 128, 0, stream>>>(prb, xth, out);
}
